// CharsLstm_9277129359701
// MI455X (gfx1250) — matrix operand order repaired and checked by reading, not compiled, not run on hardware
//
#include <hip/hip_runtime.h>
#include <hip/hip_bf16.h>

typedef __bf16 v16bf __attribute__((ext_vector_type(16)));
typedef float  v8f   __attribute__((ext_vector_type(8)));

#define B_   64
#define T_   512
#define E_   256
#define H_   1024
#define G_   4096            // 4*H
#define NBLK 32              // persistent blocks (one device barrier / step)
#define NTHR 256             // 8 wave32 waves per block -> 256 waves total

#define NCH_H (H_ / 32)          // 32 k-chunks of recurrence
#define NCH   (NCH_H + E_ / 32)  // + 8 k-chunks of input projection = 40
#define PF    5                  // software-pipeline prefetch distance (chunks)

// ---------------- workspace layout (bytes) ----------------
#define OFF_BAR   0u
#define OFF_EMB   256u                       // 256*256 bf16   = 131072
#define OFF_WIH   (OFF_EMB  + 131072u)       // 4096*256 bf16  = 2097152
#define OFF_WHH   (OFF_WIH  + 2097152u)      // 4096*1024 bf16 = 8388608
#define OFF_BIAS  (OFF_WHH  + 8388608u)      // 4096 f32       = 16384
#define OFF_HBF   (OFF_BIAS + 16384u)        // 64*1024 bf16   = 131072

// ---------------- fast transcendentals ----------------
#if __has_builtin(__builtin_amdgcn_tanhf)
__device__ __forceinline__ float fast_tanh(float x) { return __builtin_amdgcn_tanhf(x); }
#else
__device__ __forceinline__ float fast_tanh(float x) { return tanhf(x); }
#endif
__device__ __forceinline__ float sigm(float x) { return 0.5f * fast_tanh(0.5f * x) + 0.5f; }

// ---------------- scheduling fences ----------------
// mask 0x7: ALU/VALU/SALU may cross, VMEM and WMMA may NOT -> pins the
// load-ahead pipeline order so waits become partial instead of loadcnt==0.
#if __has_builtin(__builtin_amdgcn_sched_barrier)
#define SCHED_FENCE() __builtin_amdgcn_sched_barrier(0x7)
#else
#define SCHED_FENCE()
#endif

// opaque zero, regenerated each loop iteration: added to base pointers so
// LICM cannot hoist the (loop-invariant) weight loads out of the t-loop,
// while keeping kernarg provenance -> global_load (not flat_load).
__device__ __forceinline__ int opaque_zero() {
    int v = 0;
    asm volatile("" : "+v"(v));
    return v;
}

// ---------------- fragment loaders (wave32 WMMA layouts) ----------------
struct Frag32B { uint4 lo, hi; };

// A (16-bit, 16x32): lane half = lane>>4; elements 0..7 = K c..c+7 (+8*half),
// elements 8..15 = K c+16..c+23 (+8*half).  p already includes half*8.
__device__ __forceinline__ v16bf load_a_frag(const __bf16* p) {
    union { v16bf v; Frag32B f; } u;
    u.f.lo = *reinterpret_cast<const uint4*>(p);
    u.f.hi = *reinterpret_cast<const uint4*>(p + 16);
    return u.v;
}
// B (16-bit, 32x16): lane holds column n; elements 0..7 = K c..c+7 (+8*half),
// elements 8..15 = K c+16..c+23 (+8*half).  p already includes half*8.  (B = W^T: a column of B is
// a contiguous row of W.)
__device__ __forceinline__ v16bf load_b_frag(const __bf16* p) {
    union { v16bf v; Frag32B f; } u;
    u.f.lo = *reinterpret_cast<const uint4*>(p);
    u.f.hi = *reinterpret_cast<const uint4*>(p + 16);
    return u.v;
}

// ---------------- device-wide barrier (persistent kernel) ----------------
__device__ __forceinline__ void gsync(unsigned* cnt, unsigned* gen) {
    __threadfence();
    __syncthreads();
    if (threadIdx.x == 0) {
        unsigned g = __hip_atomic_load(gen, __ATOMIC_RELAXED, __HIP_MEMORY_SCOPE_AGENT);
        unsigned a = __hip_atomic_fetch_add(cnt, 1u, __ATOMIC_ACQ_REL, __HIP_MEMORY_SCOPE_AGENT);
        if (a == NBLK - 1) {
            __hip_atomic_store(cnt, 0u, __ATOMIC_RELAXED, __HIP_MEMORY_SCOPE_AGENT);
            __hip_atomic_fetch_add(gen, 1u, __ATOMIC_RELEASE, __HIP_MEMORY_SCOPE_AGENT);
        } else {
            while (__hip_atomic_load(gen, __ATOMIC_ACQUIRE, __HIP_MEMORY_SCOPE_AGENT) == g)
                __builtin_amdgcn_s_sleep(2);
        }
    }
    __syncthreads();
}

// ---------------- prep: bf16 conversions + state init ----------------
__global__ void lstm_prep(const float* __restrict__ emb,
                          const float* __restrict__ W_ih,
                          const float* __restrict__ W_hh,
                          const float* __restrict__ b_ih,
                          const float* __restrict__ b_hh,
                          const float* __restrict__ h0,
                          __bf16* __restrict__ emb_bf,
                          __bf16* __restrict__ wih_bf,
                          __bf16* __restrict__ whh_bf,
                          float*  __restrict__ bias,
                          __bf16* __restrict__ hbf,
                          unsigned* cnt, unsigned* gen) {
    const unsigned i = blockIdx.x * blockDim.x + threadIdx.x;
    if (i < (unsigned)G_ * H_)   whh_bf[i] = (__bf16)W_hh[i];
    if (i < (unsigned)G_ * E_)   wih_bf[i] = (__bf16)W_ih[i];
    if (i < 256u * 256u)         emb_bf[i] = (i < 256u) ? (__bf16)0.0f : (__bf16)emb[i]; // padding_idx=0
    if (i < (unsigned)G_)        bias[i]   = b_ih[i] + b_hh[i];
    if (i < (unsigned)B_ * H_)   hbf[i]    = (__bf16)h0[i];
    if (i == 0)                  { *cnt = 0u; *gen = 0u; }
}

// ---------------- persistent recurrent kernel ----------------
// Wave job: (mtile 0..3) x (coltile 0..63).  Each wave owns the four gate
// tiles (i,f,g,o) of one 16x16 (batch x h-unit) patch; c stays in registers
// for all 512 steps; only h (bf16) is published per step -> one barrier/step.
__global__ __launch_bounds__(NTHR)
void lstm_scan(const int* __restrict__ x,
               const __bf16* __restrict__ emb_bf,
               const __bf16* __restrict__ wih_bf,
               const __bf16* __restrict__ whh_bf,
               const float*  __restrict__ bias,
               const float*  __restrict__ c0,
               __bf16* __restrict__ hbf,
               float*  __restrict__ out,
               unsigned* cnt, unsigned* gen) {
    const int lane  = threadIdx.x & 31;
    const int wave  = threadIdx.x >> 5;
    const int w     = blockIdx.x * (NTHR / 32) + wave;   // 0..255
    const int mtile = w & 3;                             // 4 tiles cover B=64
    const int ct    = w >> 2;                            // 64 tiles cover H=1024
    const int m0 = mtile * 16, j0 = ct * 16;
    const int half = lane >> 4, lm = lane & 15;
    const int brow = m0 + lm;                            // batch row (A operand)
    const int jcol = j0 + lm;                            // h-unit column (B operand)

    const __bf16* pbih0[4];
    const __bf16* pbhh0[4];
    float bg[4];
    #pragma unroll
    for (int g = 0; g < 4; ++g) {
        const int wr = g * H_ + jcol;
        pbih0[g] = wih_bf + (size_t)wr * E_ + half * 8;
        pbhh0[g] = whh_bf + (size_t)wr * H_ + half * 8;
        bg[g]    = bias[wr];
    }
    const __bf16* pah0 = hbf + (size_t)brow * H_ + half * 8;

    // register-resident cell state: rows m0+half*8+r, column jcol
    v8f creg;
    #pragma unroll
    for (int r = 0; r < 8; ++r)
        creg[r] = c0[(size_t)(m0 + half * 8 + r) * H_ + jcol];

    for (int t = 0; t < T_; ++t) {
        // token gather issued first; its latency hides under the H-part below
        const int token = x[brow * T_ + t];

        // one opaque (always-zero) offset per iteration: keeps every frag
        // load inside the loop without losing the global address space
        const int z = opaque_zero();
        const __bf16* pa_e = emb_bf + (size_t)token * E_ + half * 8 + z;
        const __bf16* pah  = pah0 + z;
        const __bf16* pbhh[4];
        const __bf16* pbih[4];
        #pragma unroll
        for (int g = 0; g < 4; ++g) {
            pbhh[g] = pbhh0[g] + z;
            pbih[g] = pbih0[g] + z;
        }

        // unified chunk addressing: chunks 0..31 = recurrence (K=1024),
        // chunks 32..39 = input projection (K=256)
        auto a_ptr = [&](int i) {
            return (i < NCH_H) ? pah + i * 32 : pa_e + (i - NCH_H) * 32;
        };
        auto b_ptr = [&](int g, int i) {
            return (i < NCH_H) ? pbhh[g] + i * 32 : pbih[g] + (i - NCH_H) * 32;
        };

        v8f acc0 = {}, acc1 = {}, acc2 = {}, acc3 = {};

        // ---- software-pipelined GEMM: prefetch distance PF chunks ----
        v16bf abuf[PF], b0b[PF], b1b[PF], b2b[PF], b3b[PF];
        #pragma unroll
        for (int i = 0; i < PF; ++i) {
            abuf[i] = load_a_frag(a_ptr(i));
            b0b[i]  = load_b_frag(b_ptr(0, i));
            b1b[i]  = load_b_frag(b_ptr(1, i));
            b2b[i]  = load_b_frag(b_ptr(2, i));
            b3b[i]  = load_b_frag(b_ptr(3, i));
        }
        #pragma unroll
        for (int i = 0; i < NCH; ++i) {
            const int s = i % PF;
            const v16bf a  = abuf[s];
            const v16bf b0 = b0b[s], b1 = b1b[s], b2 = b2b[s], b3 = b3b[s];
            if (i + PF < NCH) {                    // static under full unroll
                abuf[s] = load_a_frag(a_ptr(i + PF));
                b0b[s]  = load_b_frag(b_ptr(0, i + PF));
                b1b[s]  = load_b_frag(b_ptr(1, i + PF));
                b2b[s]  = load_b_frag(b_ptr(2, i + PF));
                b3b[s]  = load_b_frag(b_ptr(3, i + PF));
            }
            SCHED_FENCE();   // loads(i+PF) must issue before wmma(i)
            acc0 = __builtin_amdgcn_wmma_f32_16x16x32_bf16(false, a, false, b0,
                                                           (short)0, acc0, false, false);
            acc1 = __builtin_amdgcn_wmma_f32_16x16x32_bf16(false, a, false, b1,
                                                           (short)0, acc1, false, false);
            acc2 = __builtin_amdgcn_wmma_f32_16x16x32_bf16(false, a, false, b2,
                                                           (short)0, acc2, false, false);
            acc3 = __builtin_amdgcn_wmma_f32_16x16x32_bf16(false, a, false, b3,
                                                           (short)0, acc3, false, false);
            SCHED_FENCE();   // wmma(i) must issue before loads(i+PF+1)
        }

        // ---- in-register LSTM pointwise update (all four gates local) ----
        #pragma unroll
        for (int r = 0; r < 8; ++r) {
            const float iv = sigm (acc0[r] + bg[0]);
            const float fv = sigm (acc1[r] + bg[1]);
            const float gv = fast_tanh(acc2[r] + bg[2]);
            const float ov = sigm (acc3[r] + bg[3]);
            const float cv = fv * creg[r] + iv * gv;
            creg[r] = cv;
            const float hv = ov * fast_tanh(cv);
            const size_t idx = (size_t)(m0 + half * 8 + r) * H_ + jcol;
            hbf[idx] = (__bf16)hv;
            if (t == T_ - 1) out[idx] = hv;
        }

        gsync(cnt, gen);   // publish h_bf for every wave's next-step A loads
    }
}

// ---------------- host ----------------
extern "C" void kernel_launch(void* const* d_in, const int* in_sizes, int n_in,
                              void* d_out, int out_size, void* d_ws, size_t ws_size,
                              hipStream_t stream) {
    const int*   x    = (const int*)  d_in[0];
    const float* emb  = (const float*)d_in[1];
    const float* W_ih = (const float*)d_in[2];
    const float* W_hh = (const float*)d_in[3];
    const float* b_ih = (const float*)d_in[4];
    const float* b_hh = (const float*)d_in[5];
    const float* h0   = (const float*)d_in[6];
    const float* c0   = (const float*)d_in[7];
    float* out = (float*)d_out;

    char* ws = (char*)d_ws;
    unsigned* cnt   = (unsigned*)(ws + OFF_BAR);
    unsigned* gen   = (unsigned*)(ws + OFF_BAR + 8);
    __bf16* emb_bf  = (__bf16*)(ws + OFF_EMB);
    __bf16* wih_bf  = (__bf16*)(ws + OFF_WIH);
    __bf16* whh_bf  = (__bf16*)(ws + OFF_WHH);
    float*  bias    = (float*) (ws + OFF_BIAS);
    __bf16* hbf     = (__bf16*)(ws + OFF_HBF);

    const int prep_n   = G_ * H_;                   // largest conversion: W_hh
    const int prep_blk = (prep_n + NTHR - 1) / NTHR;
    lstm_prep<<<prep_blk, NTHR, 0, stream>>>(emb, W_ih, W_hh, b_ih, b_hh, h0,
                                             emb_bf, wih_bf, whh_bf, bias,
                                             hbf, cnt, gen);

    lstm_scan<<<NBLK, NTHR, 0, stream>>>(x, emb_bf, wih_bf, whh_bf, bias, c0,
                                         hbf, out, cnt, gen);
}
